// ToMeLayer_65687229825445
// MI455X (gfx1250) — matrix operand order repaired and checked by reading, not compiled, not run on hardware
//
#include <hip/hip_runtime.h>
#include <hip/hip_bf16.h>
#include <stdint.h>

#define B_  8
#define T_  4096
#define D_  512
#define R_  1024
#define HT_ (T_ / 2)   // 2048 src rows, 2048 dst rows per batch

typedef __attribute__((ext_vector_type(16))) __bf16 v16bf;
typedef __attribute__((ext_vector_type(8)))  __bf16 v8bf;
typedef __attribute__((ext_vector_type(8)))  float  v8f;
typedef unsigned int u32x4 __attribute__((ext_vector_type(4)));
typedef int          i32x8 __attribute__((ext_vector_type(8)));
typedef int          i32x4 __attribute__((ext_vector_type(4)));

__device__ __forceinline__ unsigned short f32_to_bf16(float f) {
    unsigned u = __float_as_uint(f);
    unsigned r = u + 0x7FFFu + ((u >> 16) & 1u);   // round-to-nearest-even
    return (unsigned short)(r >> 16);
}

// ---------------------------------------------------------------------------
// Init partner / addsrc maps to zero (graph-replay safe: rewritten every call)
// ---------------------------------------------------------------------------
__global__ void k_init(int* __restrict__ partner, int* __restrict__ addsrc) {
    int i = blockIdx.x * blockDim.x + threadIdx.x;
    if (i < B_ * T_) { partner[i] = 0; addsrc[i] = 0; }
}

// ---------------------------------------------------------------------------
// L2-normalize rows of x, split into src (odd t) / dst (even t), store bf16
// ---------------------------------------------------------------------------
__global__ __launch_bounds__(128)
void k_normalize(const float* __restrict__ x,
                 unsigned short* __restrict__ srcn,
                 unsigned short* __restrict__ dstn) {
    int wave = (blockIdx.x * blockDim.x + threadIdx.x) >> 5;
    int lane = threadIdx.x & 31;
    int b = wave / T_;
    int t = wave % T_;

    const float4* r4 = (const float4*)(x + (size_t)wave * D_ + lane * 16);
    float vals[16];
    float ss = 0.f;
    #pragma unroll
    for (int i = 0; i < 4; ++i) {
        float4 v = r4[i];
        vals[4*i+0] = v.x; vals[4*i+1] = v.y; vals[4*i+2] = v.z; vals[4*i+3] = v.w;
        ss += v.x*v.x + v.y*v.y + v.z*v.z + v.w*v.w;
    }
    #pragma unroll
    for (int m = 1; m < 32; m <<= 1) ss += __shfl_xor(ss, m, 32);
    float scale = 1.0f / fmaxf(sqrtf(ss), 1e-12f);

    alignas(16) unsigned short ob[16];
    #pragma unroll
    for (int i = 0; i < 16; ++i) ob[i] = f32_to_bf16(vals[i] * scale);

    unsigned short* row = (t & 1)
        ? srcn + ((size_t)b * HT_ + (t >> 1)) * D_
        : dstn + ((size_t)b * HT_ + (t >> 1)) * D_;
    uint4* o = (uint4*)(row + lane * 16);
    o[0] = *(const uint4*)&ob[0];
    o[1] = *(const uint4*)&ob[8];
}

// ---------------------------------------------------------------------------
// Pre-transpose W into bf16 Wt[n][k]  (B-operand K-runs contiguous)
// ---------------------------------------------------------------------------
__global__ void k_wprep(const float* __restrict__ W, unsigned short* __restrict__ Wt) {
    int i = blockIdx.x * blockDim.x + threadIdx.x;
    if (i >= D_ * D_) return;
    int n = i / D_, k = i % D_;
    Wt[n * D_ + k] = f32_to_bf16(W[k * D_ + n]);
}

// ---------------------------------------------------------------------------
// Scores GEMM with fused row-max/argmax. dst tiles staged memory->LDS with
// CDNA5 async-to-LDS (ASYNCcnt) and double buffering: fetch of tile j+1
// overlaps the 16-WMMA chain on tile j.
// ---------------------------------------------------------------------------
__global__ __launch_bounds__(128)
void k_scores(const unsigned short* __restrict__ srcn,
              const unsigned short* __restrict__ dstn,
              float* __restrict__ best_scores,
              int*   __restrict__ best_local) {
    __shared__ alignas(32) unsigned short Bt[2][16 * D_];   // 2 x 16 KB

    int b    = blockIdx.y;
    int wv   = threadIdx.x >> 5;
    int lane = threadIdx.x & 31;
    int tile = blockIdx.x * 4 + wv;       // src tile 0..127
    int srcbase = tile * 16;
    int mrow = lane & 15;
    int half = lane >> 4;

    const unsigned short* db = dstn + (size_t)b * HT_ * D_;  // uniform SGPR base

    // A operand: lane l holds row (l&15); K = h*16..+15, h=l>>4
    const unsigned short* abase = srcn + ((size_t)b * HT_ + srcbase + mrow) * D_;
    v16bf A[16];
    #pragma unroll
    for (int c = 0; c < 16; ++c) {
        int k0 = c * 32 + half * 16;
        v8bf lo = *(const v8bf*)(abase + k0);
        v8bf hi = *(const v8bf*)(abase + k0 + 8);
        #pragma unroll
        for (int i = 0; i < 8; ++i) { A[c][i] = lo[i]; A[c][8 + i] = hi[i]; }
    }

    // async memory->LDS tile fetch: 128 threads x 128B = 16KB, no VGPR staging.
    // Instruction IOFFSET is added to BOTH the LDS and global addresses.
    int flat = threadIdx.x * 64;                         // half-elements
    unsigned voff_base = (unsigned)(((flat >> 9) * D_ + (flat & 511)) * 2);
    auto issue_tile = [&](int j, int buf) {
        unsigned lds_a = (unsigned)(uintptr_t)&Bt[buf][flat];
        unsigned voff  = voff_base + (unsigned)(j * 16 * D_ * 2);
        asm volatile(
            "global_load_async_to_lds_b128 %0, %1, %2 offset:0\n\t"
            "global_load_async_to_lds_b128 %0, %1, %2 offset:16\n\t"
            "global_load_async_to_lds_b128 %0, %1, %2 offset:32\n\t"
            "global_load_async_to_lds_b128 %0, %1, %2 offset:48\n\t"
            "global_load_async_to_lds_b128 %0, %1, %2 offset:64\n\t"
            "global_load_async_to_lds_b128 %0, %1, %2 offset:80\n\t"
            "global_load_async_to_lds_b128 %0, %1, %2 offset:96\n\t"
            "global_load_async_to_lds_b128 %0, %1, %2 offset:112\n\t"
            :: "v"(lds_a), "v"(voff), "s"(db) : "memory");
    };

    float best[8]; int bidx[8];
    #pragma unroll
    for (int v = 0; v < 8; ++v) { best[v] = -3.4e38f; bidx[v] = 0; }

    issue_tile(0, 0);                                    // prologue
    for (int j = 0; j < HT_ / 16; ++j) {
        int cur = j & 1;
        asm volatile("s_wait_asynccnt 0x0" ::: "memory"); // my tile-j loads done
        __syncthreads();   // everyone's tile j visible; buf cur^1 reads (tile j-1) retired
        if (j + 1 < HT_ / 16) issue_tile(j + 1, cur ^ 1); // overlap with compute

        v8f c = {0.f,0.f,0.f,0.f,0.f,0.f,0.f,0.f};
        #pragma unroll
        for (int ck = 0; ck < 16; ++ck) {
            v16bf bv = *(const v16bf*)(&Bt[cur][mrow * D_ + ck * 32 + half * 16]);
            c = __builtin_amdgcn_wmma_f32_16x16x32_bf16(
                    false, A[ck], false, bv, (short)0, c, false, false);
        }

        // fused row-max/argmax: row m = v + 8*half, n = lane&15
        #pragma unroll
        for (int v = 0; v < 8; ++v) {
            float val = c[v];
            int   idx = j * 16 + mrow;
            #pragma unroll
            for (int m = 1; m < 16; m <<= 1) {
                float ov = __shfl_xor(val, m, 32);
                int   oi = __shfl_xor(idx, m, 32);
                if (ov > val || (ov == val && oi < idx)) { val = ov; idx = oi; }
            }
            if (val > best[v]) { best[v] = val; bidx[v] = idx; }   // strict >: lowest idx wins
        }
    }

    if (mrow == 0) {
        #pragma unroll
        for (int v = 0; v < 8; ++v) {
            int row = srcbase + half * 8 + v;
            best_scores[b * HT_ + row] = best[v];
            best_local [b * HT_ + row] = bidx[v];
        }
    }
}

// ---------------------------------------------------------------------------
// Exact top-R select per batch (4-pass MSB radix select; ties -> lowest index)
// ---------------------------------------------------------------------------
__global__ __launch_bounds__(1024)
void k_topk(const float* __restrict__ best_scores, const int* __restrict__ best_local,
            int* __restrict__ partner, int* __restrict__ addsrc) {
    __shared__ unsigned keys[HT_];
    __shared__ unsigned scanbuf[HT_];
    __shared__ unsigned hist[256];
    __shared__ unsigned sh_prefix, sh_rank;

    int b = blockIdx.x;
    int tid = threadIdx.x;

    for (int i = tid; i < HT_; i += 1024) {
        unsigned u = __float_as_uint(best_scores[b * HT_ + i]);
        keys[i] = (u & 0x80000000u) ? ~u : (u | 0x80000000u);
    }
    if (tid == 0) { sh_prefix = 0u; sh_rank = R_; }
    __syncthreads();

    for (int pass = 0; pass < 4; ++pass) {
        int shift = 24 - 8 * pass;
        unsigned maskhi = (pass == 0) ? 0u : (0xFFFFFFFFu << (shift + 8));
        if (tid < 256) hist[tid] = 0u;
        __syncthreads();
        unsigned prefix = sh_prefix;
        for (int i = tid; i < HT_; i += 1024) {
            unsigned k = keys[i];
            if ((k & maskhi) == prefix) atomicAdd(&hist[(k >> shift) & 0xFFu], 1u);
        }
        __syncthreads();
        if (tid == 0) {
            unsigned cum = 0, rank = sh_rank, chosen = 0;
            for (int bin = 255; bin >= 0; --bin) {
                unsigned c = hist[bin];
                if (cum + c >= rank) { chosen = (unsigned)bin; rank -= cum; break; }
                cum += c;
            }
            sh_rank = rank;
            sh_prefix = prefix | (chosen << shift);
        }
        __syncthreads();
    }
    unsigned thr = sh_prefix;
    unsigned eneed = sh_rank;

    for (int i = tid; i < HT_; i += 1024) scanbuf[i] = (keys[i] == thr) ? 1u : 0u;
    __syncthreads();
    if (tid == 0) {
        unsigned run = 0;
        for (int i = 0; i < HT_; ++i) { unsigned t = scanbuf[i]; scanbuf[i] = run; run += t; }
    }
    __syncthreads();

    for (int i = tid; i < HT_; i += 1024) {
        unsigned k = keys[i];
        bool sel = (k > thr) || (k == thr && scanbuf[i] < eneed);
        if (sel) {
            int dl   = best_local[b * HT_ + i];
            int spos = 2 * i + 1;
            int dpos = 2 * dl;
            partner[b * T_ + spos] = dpos + 1;   // out[spos] <- out[dpos]
            addsrc [b * T_ + dpos] = spos + 1;   // x_merged[dpos] = (x[dpos]+x[spos])/2
        }
    }
}

// ---------------------------------------------------------------------------
// Build bf16 A matrix for the projection GEMM, folding the merge in.
// ---------------------------------------------------------------------------
__global__ __launch_bounds__(128)
void k_buildA(const float* __restrict__ x, const int* __restrict__ addsrc,
              unsigned short* __restrict__ Abf) {
    int wave = (blockIdx.x * blockDim.x + threadIdx.x) >> 5;
    int lane = threadIdx.x & 31;
    int b = wave / T_;

    float vals[16];
    const float4* r4 = (const float4*)(x + (size_t)wave * D_ + lane * 16);
    #pragma unroll
    for (int i = 0; i < 4; ++i) {
        float4 v = r4[i];
        vals[4*i+0]=v.x; vals[4*i+1]=v.y; vals[4*i+2]=v.z; vals[4*i+3]=v.w;
    }
    int a = addsrc[wave];                 // wave-uniform
    if (a) {
        const float4* s4 = (const float4*)(x + ((size_t)b * T_ + (a - 1)) * D_ + lane * 16);
        #pragma unroll
        for (int i = 0; i < 4; ++i) {
            float4 v = s4[i];
            vals[4*i+0]=(vals[4*i+0]+v.x)*0.5f; vals[4*i+1]=(vals[4*i+1]+v.y)*0.5f;
            vals[4*i+2]=(vals[4*i+2]+v.z)*0.5f; vals[4*i+3]=(vals[4*i+3]+v.w)*0.5f;
        }
    }
    alignas(16) unsigned short ob[16];
    #pragma unroll
    for (int i = 0; i < 16; ++i) ob[i] = f32_to_bf16(vals[i]);
    uint4* o = (uint4*)(Abf + (size_t)wave * D_ + lane * 16);
    o[0] = *(const uint4*)&ob[0];
    o[1] = *(const uint4*)&ob[8];
}

// ---------------------------------------------------------------------------
// Projection GEMM: out = A(bf16) @ W + b. Wt tiles staged via the Tensor
// Data Mover (one descriptor-driven DMA per tile, TENSORcnt), double-buffered.
// ---------------------------------------------------------------------------
__global__ __launch_bounds__(128)
void k_gemm2(const unsigned short* __restrict__ Abf,
             const unsigned short* __restrict__ Wt,
             const float* __restrict__ bias,
             float* __restrict__ out) {
    __shared__ alignas(32) unsigned short Bt[2][16 * D_];   // 2 x 16 KB

    int wv   = threadIdx.x >> 5;
    int lane = threadIdx.x & 31;
    int mrow = lane & 15;
    int half = lane >> 4;
    int rowbase = blockIdx.x * 64 + wv * 16;

    const unsigned short* abase = Abf + ((size_t)(rowbase + mrow)) * D_;
    v16bf A[16];
    #pragma unroll
    for (int c = 0; c < 16; ++c) {
        int k0 = c * 32 + half * 16;
        v8bf lo = *(const v8bf*)(abase + k0);
        v8bf hi = *(const v8bf*)(abase + k0 + 8);
        #pragma unroll
        for (int i = 0; i < 8; ++i) { A[c][i] = lo[i]; A[c][8 + i] = hi[i]; }
    }

    // TDM descriptor for one 16x512-bf16 Wt tile, viewed as 8-byte elements:
    // 2D tile 128 x 16, row stride 128 elements (contiguous).
    auto tdm_issue = [&](int nt, int buf) {
        unsigned long long ga = (unsigned long long)(uintptr_t)Wt
                              + (unsigned long long)nt * 16ull * D_ * 2ull;
        unsigned lds_a = (unsigned)(uintptr_t)&Bt[buf][0];
        u32x4 g0;
        g0[0] = 1u;                                   // count=1, user descriptor
        g0[1] = lds_a;                                // lds_addr
        g0[2] = (unsigned)ga;                         // global_addr[31:0]
        g0[3] = (unsigned)((ga >> 32) & 0x1FFFFFFu) | (2u << 30);  // ga[56:32], type=2
        i32x8 g1;
        g1[0] = (3 << 16);            // workgroup_mask=0, data_size=3 (8B)
        g1[1] = (128 << 16);          // tensor_dim0[15:0]=128  (bits 79:48)
        g1[2] = (512 << 16);          // tensor_dim0 hi=0, tensor_dim1[15:0]=512
        g1[3] = (128 << 16);          // tensor_dim1 hi=0, tile_dim0=128
        g1[4] = 16;                   // tile_dim1=16, tile_dim2=0
        g1[5] = 128;                  // tensor_dim0_stride[31:0]=128
        g1[6] = 0;                    // stride hi / dim1_stride lo
        g1[7] = 0;
        i32x4 z4 = {0, 0, 0, 0};
        i32x8 z8 = {0, 0, 0, 0, 0, 0, 0, 0};
        // 6-arg toolchain variant: (g0, g1, g2, g3, g4, cpol)
        __builtin_amdgcn_tensor_load_to_lds(g0, g1, z4, z4, z8, 0);
    };

    const int NT = D_ / 16;
    if (threadIdx.x < 32) tdm_issue(0, 0);            // prologue (wave 0 drives TDM)

    for (int nt = 0; nt < NT; ++nt) {
        int cur = nt & 1;
        __syncthreads();          // all reads of buf cur^1 (tile nt-1) retired
        if (threadIdx.x < 32) {
            if (nt + 1 < NT) {
                tdm_issue(nt + 1, cur ^ 1);           // keep one DMA in flight
                __builtin_amdgcn_s_wait_tensorcnt(1); // oldest (tile nt) complete
            } else {
                __builtin_amdgcn_s_wait_tensorcnt(0);
            }
        }
        __syncthreads();          // tile nt visible to the whole workgroup

        v8f c = {0.f,0.f,0.f,0.f,0.f,0.f,0.f,0.f};
        #pragma unroll
        for (int ck = 0; ck < 16; ++ck) {
            v16bf bv = *(const v16bf*)(&Bt[cur][mrow * D_ + ck * 32 + half * 16]);
            c = __builtin_amdgcn_wmma_f32_16x16x32_bf16(
                    false, A[ck], false, bv, (short)0, c, false, false);
        }
        int n = nt * 16 + mrow;
        float bb = bias[n];
        #pragma unroll
        for (int v = 0; v < 8; ++v) {
            int g = rowbase + half * 8 + v;
            out[(size_t)g * D_ + n] = c[v] + bb;
        }
    }
}

// ---------------------------------------------------------------------------
// Fix-up: merged-away src rows copy their partner (dst) row.
// ---------------------------------------------------------------------------
__global__ __launch_bounds__(128)
void k_fixup(const int* __restrict__ partner, float* __restrict__ out) {
    int wave = (blockIdx.x * blockDim.x + threadIdx.x) >> 5;
    int lane = threadIdx.x & 31;
    int p = partner[wave];                // wave-uniform
    if (p == 0) return;
    int b = wave / T_;
    const float4* s = (const float4*)(out + ((size_t)(b * T_ + p - 1)) * D_ + lane * 16);
    float4*       d = (float4*)(out + (size_t)wave * D_ + lane * 16);
    #pragma unroll
    for (int i = 0; i < 4; ++i) d[i] = s[i];
}

// ---------------------------------------------------------------------------
extern "C" void kernel_launch(void* const* d_in, const int* in_sizes, int n_in,
                              void* d_out, int out_size, void* d_ws, size_t ws_size,
                              hipStream_t stream) {
    (void)in_sizes; (void)n_in; (void)out_size; (void)ws_size;
    const float* x    = (const float*)d_in[0];
    const float* W    = (const float*)d_in[1];
    const float* bias = (const float*)d_in[2];
    float* out = (float*)d_out;

    char* ws = (char*)d_ws;
    unsigned short* srcn = (unsigned short*)(ws);
    unsigned short* dstn = (unsigned short*)(ws + (size_t)16 * 1024 * 1024);
    unsigned short* Abf  = (unsigned short*)(ws);    // reuses src/dst region, stream-ordered
    unsigned short* Wt   = (unsigned short*)(ws + (size_t)32 * 1024 * 1024);
    float* bscores = (float*)(ws + 33554432 + 524288);
    int*   blocal  = (int*)  (ws + 33554432 + 524288 + 65536);
    int*   partner = (int*)  (ws + 33554432 + 524288 + 2 * 65536);
    int*   addsrc  = (int*)  (ws + 33554432 + 524288 + 2 * 65536 + 131072);

    k_init     <<<(B_ * T_ + 255) / 256, 256, 0, stream>>>(partner, addsrc);
    k_normalize<<<B_ * T_ / 4, 128, 0, stream>>>(x, srcn, dstn);
    k_wprep    <<<(D_ * D_ + 255) / 256, 256, 0, stream>>>(W, Wt);

    dim3 gs(HT_ / 16 / 4, B_);   // 32 x 8 blocks, 4 waves each
    k_scores   <<<gs, 128, 0, stream>>>(srcn, dstn, bscores, blocal);
    k_topk     <<<B_, 1024, 0, stream>>>(bscores, blocal, partner, addsrc);
    k_buildA   <<<B_ * T_ / 4, 128, 0, stream>>>(x, addsrc, Abf);
    k_gemm2    <<<B_ * T_ / 64, 128, 0, stream>>>(Abf, Wt, bias, out);
    k_fixup    <<<B_ * T_ / 4, 128, 0, stream>>>(partner, out);
}
